// VarianceAdapter_23880018166300
// MI455X (gfx1250) — matrix operand order repaired and checked by reading, not compiled, not run on hardware
//
#include <hip/hip_runtime.h>
#include <hip/hip_bf16.h>

// ---------------------------------------------------------------------------
// Problem constants (match reference)
// ---------------------------------------------------------------------------
#define BB 16
#define LPH 256
#define DD 256
#define MM 2048
#define NBINS 256

typedef __attribute__((ext_vector_type(16))) __bf16 v16bf;
typedef __attribute__((ext_vector_type(8)))  float  v8f;

union ABFrag { v16bf v; uint4 u[2]; };

// ---------------------------------------------------------------------------
// Weight transpose + bf16 convert: w[co][ci][k] (f32) -> wt[k][co][ci] (bf16)
// N = 3*256*256 = 196608
// ---------------------------------------------------------------------------
__global__ __launch_bounds__(256)
void cvt_w_kernel(const float* __restrict__ w, __hip_bfloat16* __restrict__ wt) {
    int i  = blockIdx.x * 256 + threadIdx.x;     // dst flat index
    int ci = i & 255;
    int co = (i >> 8) & 255;
    int k  = i >> 16;
    wt[i] = __float2bfloat16(w[(co * 256 + ci) * 3 + k]);
}

// f32 -> bf16 bulk convert
__global__ __launch_bounds__(256)
void cvt_x_kernel(const float* __restrict__ x, __hip_bfloat16* __restrict__ y, int n) {
    int i = blockIdx.x * 256 + threadIdx.x;
    if (i < n) y[i] = __float2bfloat16(x[i]);
}

// ---------------------------------------------------------------------------
// Per-batch inclusive scan of durations -> ends[b][l]
// ---------------------------------------------------------------------------
__global__ __launch_bounds__(256)
void scan_kernel(const int* __restrict__ dur, int* __restrict__ ends) {
    __shared__ int s[256];
    int b = blockIdx.x, t = threadIdx.x;
    s[t] = dur[b * 256 + t];
    __syncthreads();
    for (int off = 1; off < 256; off <<= 1) {
        int u = (t >= off) ? s[t - off] : 0;
        __syncthreads();
        s[t] += u;
        __syncthreads();
    }
    ends[b * 256 + t] = s[t];
}

// ---------------------------------------------------------------------------
// Length regulate: frame t copies phoneme j with starts[j] <= t < ends[j].
// One block per (b, t); 256 threads over channels.
// ---------------------------------------------------------------------------
__global__ __launch_bounds__(256)
void lr_expand_kernel(const float* __restrict__ enc, const int* __restrict__ ends,
                      float* __restrict__ outf, __hip_bfloat16* __restrict__ outb) {
    int bm = blockIdx.x;
    int b  = bm >> 11;          // / 2048
    int t  = bm & (MM - 1);
    int tid = threadIdx.x;
    const int* e = ends + b * 256;
    int lo = 0, hi = 256;
    while (lo < hi) {           // first j with ends[j] > t
        int mid = (lo + hi) >> 1;
        if (e[mid] <= t) lo = mid + 1; else hi = mid;
    }
    float v = 0.0f;
    if (lo < 256) v = enc[((size_t)b * 256 + lo) * 256 + tid];
    size_t o = (size_t)bm * 256 + tid;
    outf[o] = v;
    outb[o] = __float2bfloat16(v);
}

// ---------------------------------------------------------------------------
// Fused Conv1d(K=3, pad=1, 256->256) + LayerNorm(channel) + ReLU via WMMA.
// Block = 256 threads (8 waves); computes 16 time-rows x 256 channels.
// Each wave owns two 16x16 column tiles; 3(k) x 8(ci-chunks) x 2 WMMAs.
// X, Wt in bf16; Wt layout [k][co][ci]; H out bf16.
// ---------------------------------------------------------------------------
__global__ __launch_bounds__(256)
void conv_ln_relu_kernel(const __hip_bfloat16* __restrict__ X,
                         const __hip_bfloat16* __restrict__ Wt,
                         const float* __restrict__ bias,
                         const float* __restrict__ gamma,
                         const float* __restrict__ beta,
                         __hip_bfloat16* __restrict__ H,
                         int T) {
    __shared__ float smem[16 * 256];
    __shared__ float redA[256];
    __shared__ float redB[256];
    __shared__ float s_mean[16];
    __shared__ float s_rstd[16];

    const int tiles_t = T >> 4;
    const int b    = blockIdx.x / tiles_t;
    const int trow = (blockIdx.x % tiles_t) << 4;
    const int tid  = threadIdx.x;
    const int wv   = tid >> 5;
    const int lane = tid & 31;
    const int lane15 = lane & 15;
    const bool hiHalf = (lane >= 16);
    const int co0 = wv << 5;            // this wave: columns [co0, co0+32)

    v8f acc0 = {};
    v8f acc1 = {};

    const __hip_bfloat16* xb = X + (size_t)b * T * 256;

    for (int k = 0; k < 3; ++k) {
        const int row = trow + lane15 + k - 1;
        const bool rok = (row >= 0) && (row < T);
        const __hip_bfloat16* wk = Wt + (size_t)k * 65536;
        for (int ci0 = 0; ci0 < 256; ci0 += 32) {
            ABFrag a, b0, b1;
            // A fragment: 16x32 bf16 tile of X (row = time, col = ci), padded rows -> 0
            const int c0 = ci0 + (hiHalf ? 16 : 0);
            if (rok) {
                const __hip_bfloat16* xr = xb + (size_t)row * 256 + c0;
                a.u[0] = *(const uint4*)(xr);
                a.u[1] = *(const uint4*)(xr + 8);
            } else {
                a.u[0] = make_uint4(0u, 0u, 0u, 0u);
                a.u[1] = make_uint4(0u, 0u, 0u, 0u);
            }
            // B fragments: 32x16 bf16 tiles of Wt[k] (K = ci, N = co), contiguous per lane
            const int cik = ci0 + (hiHalf ? 16 : 0);
            const __hip_bfloat16* wp0 = wk + (size_t)(co0 + lane15) * 256 + cik;
            b0.u[0] = *(const uint4*)(wp0);
            b0.u[1] = *(const uint4*)(wp0 + 8);
            const __hip_bfloat16* wp1 = wk + (size_t)(co0 + 16 + lane15) * 256 + cik;
            b1.u[0] = *(const uint4*)(wp1);
            b1.u[1] = *(const uint4*)(wp1 + 8);

            acc0 = __builtin_amdgcn_wmma_f32_16x16x32_bf16(false, a.v, false, b0.v,
                                                           (short)0, acc0, false, false);
            acc1 = __builtin_amdgcn_wmma_f32_16x16x32_bf16(false, a.v, false, b1.v,
                                                           (short)0, acc1, false, false);
        }
    }

    // Spill accumulators (+conv bias) to LDS: smem[m][co]
    {
        const int m0  = hiHalf ? 8 : 0;
        const int c_a = co0 + lane15;
        const int c_b = co0 + 16 + lane15;
        const float bia = bias[c_a];
        const float bib = bias[c_b];
#pragma unroll
        for (int v = 0; v < 8; ++v) {
            smem[(m0 + v) * 256 + c_a] = acc0[v] + bia;
            smem[(m0 + v) * 256 + c_b] = acc1[v] + bib;
        }
    }
    __syncthreads();

    // LayerNorm over 256 channels per row; 16 threads per row
    const int r   = tid >> 4;
    const int sub = tid & 15;
    float s = 0.0f, ss = 0.0f;
#pragma unroll
    for (int i = 0; i < 16; ++i) {
        float x = smem[r * 256 + sub * 16 + i];
        s += x; ss += x * x;
    }
    redA[r * 16 + sub] = s;
    redB[r * 16 + sub] = ss;
    __syncthreads();
    if (tid < 16) {
        float su = 0.0f, sq = 0.0f;
#pragma unroll
        for (int i = 0; i < 16; ++i) { su += redA[tid * 16 + i]; sq += redB[tid * 16 + i]; }
        float mean = su * (1.0f / 256.0f);
        float var  = sq * (1.0f / 256.0f) - mean * mean;
        s_mean[tid] = mean;
        s_rstd[tid] = rsqrtf(var + 1e-5f);
    }
    __syncthreads();

    const float mean = s_mean[r];
    const float rstd = s_rstd[r];
    __hip_bfloat16* hrow = H + ((size_t)b * T + trow + r) * 256;
#pragma unroll
    for (int i = 0; i < 16; ++i) {
        int c = sub * 16 + i;
        float x = smem[r * 256 + c];
        float y = (x - mean) * rstd * gamma[c] + beta[c];
        y = fmaxf(y, 0.0f);
        hrow[c] = __float2bfloat16(y);
    }
}

// ---------------------------------------------------------------------------
// Final Linear(256 -> 1) + ReLU. One wave per row.
// ---------------------------------------------------------------------------
__global__ __launch_bounds__(256)
void linear_relu_kernel(const __hip_bfloat16* __restrict__ H,
                        const float* __restrict__ wl, const float* __restrict__ bl,
                        float* __restrict__ out) {
    int row  = blockIdx.x * 8 + (threadIdx.x >> 5);
    int lane = threadIdx.x & 31;
    const __hip_bfloat16* hr = H + (size_t)row * 256;
    float s = 0.0f;
#pragma unroll
    for (int i = 0; i < 8; ++i) {
        int c = lane * 8 + i;
        s += __bfloat162float(hr[c]) * wl[c];
    }
    for (int off = 16; off > 0; off >>= 1) s += __shfl_down(s, off, 32);
    if (lane == 0) out[row] = fmaxf(s + bl[0], 0.0f);
}

// ---------------------------------------------------------------------------
// Bucketize (lower_bound over linspace bins) + embedding add, in place.
// ---------------------------------------------------------------------------
__device__ __forceinline__ int bucket_idx(float x, float b1) {
    // bins[i] = 0 + i * b1/255 ; first i with bins[i] >= x ; clamp to 255
    const float step = b1 * (1.0f / 255.0f);
    int lo = 0, hi = 256;
    while (lo < hi) {
        int mid = (lo + hi) >> 1;
        float bv = fmaf((float)mid, step, 0.0f);
        if (bv < x) lo = mid + 1; else hi = mid;
    }
    return (lo > 255) ? 255 : lo;
}

__global__ __launch_bounds__(256)
void combine_kernel(const float* __restrict__ etgt, const float* __restrict__ ptgt,
                    const float* __restrict__ eemb, const float* __restrict__ pemb,
                    float* __restrict__ comb) {
    int bm  = blockIdx.x;
    int tid = threadIdx.x;
    float xe = log1pf(etgt[bm]);
    float xp = log1pf(ptgt[bm]);
    int ie = bucket_idx(xe, logf(102.0f));   // log(E_MAX + 2)
    int ip = bucket_idx(xp, logf(402.0f));   // log(P_MAX + 2)
    size_t o = (size_t)bm * 256 + tid;
    comb[o] = comb[o] + eemb[ie * 256 + tid] + pemb[ip * 256 + tid];
}

// ---------------------------------------------------------------------------
// Host-side orchestration
// ---------------------------------------------------------------------------
extern "C" void kernel_launch(void* const* d_in, const int* in_sizes, int n_in,
                              void* d_out, int out_size, void* d_ws, size_t ws_size,
                              hipStream_t stream) {
    (void)in_sizes; (void)n_in; (void)out_size; (void)ws_size;

    const float* enc  = (const float*)d_in[0];
    const int*   ltgt = (const int*)d_in[1];
    const float* etgt = (const float*)d_in[2];
    const float* ptgt = (const float*)d_in[3];
    // d_in[4] = mel_max_length (compile-time M)
    const float* dp[10]; const float* ep[10]; const float* pp[10];
    for (int i = 0; i < 10; ++i) {
        dp[i] = (const float*)d_in[5 + i];
        ep[i] = (const float*)d_in[15 + i];
        pp[i] = (const float*)d_in[25 + i];
    }
    const float* eemb = (const float*)d_in[35];
    const float* pemb = (const float*)d_in[36];

    // d_out layout: combined[16,2048,256] | dur[16,256] | ener[16,2048] | pitch[16,2048]
    float* comb = (float*)d_out;
    float* durp = comb + (size_t)BB * MM * DD;
    float* enp  = durp + (size_t)BB * LPH;
    float* pip  = enp  + (size_t)BB * MM;

    // Workspace carve-up
    char* ws = (char*)d_ws;
    size_t off = 0;
    auto carve = [&](size_t bytes) -> void* {
        void* p = ws + off;
        off = (off + bytes + 255) & ~(size_t)255;
        return p;
    };
    int*             ends   = (int*)carve((size_t)BB * LPH * 4);
    __hip_bfloat16*  encbf  = (__hip_bfloat16*)carve((size_t)BB * LPH * DD * 2);
    __hip_bfloat16*  outbf  = (__hip_bfloat16*)carve((size_t)BB * MM * DD * 2);
    __hip_bfloat16*  wt[6];
    for (int i = 0; i < 6; ++i) wt[i] = (__hip_bfloat16*)carve((size_t)3 * 256 * 256 * 2);
    __hip_bfloat16*  h1     = (__hip_bfloat16*)carve((size_t)BB * MM * DD * 2);
    __hip_bfloat16*  h2     = (__hip_bfloat16*)carve((size_t)BB * MM * DD * 2);

    // 1) Convert conv weights to bf16 [k][co][ci]
    hipLaunchKernelGGL(cvt_w_kernel, dim3(768), dim3(256), 0, stream, dp[0], wt[0]);
    hipLaunchKernelGGL(cvt_w_kernel, dim3(768), dim3(256), 0, stream, dp[4], wt[1]);
    hipLaunchKernelGGL(cvt_w_kernel, dim3(768), dim3(256), 0, stream, ep[0], wt[2]);
    hipLaunchKernelGGL(cvt_w_kernel, dim3(768), dim3(256), 0, stream, ep[4], wt[3]);
    hipLaunchKernelGGL(cvt_w_kernel, dim3(768), dim3(256), 0, stream, pp[0], wt[4]);
    hipLaunchKernelGGL(cvt_w_kernel, dim3(768), dim3(256), 0, stream, pp[4], wt[5]);

    // 2) enc -> bf16
    hipLaunchKernelGGL(cvt_x_kernel, dim3((BB * LPH * DD) / 256), dim3(256), 0, stream,
                       enc, encbf, BB * LPH * DD);

    // 3) duration scan + length regulate (writes f32 into combined region + bf16 copy)
    hipLaunchKernelGGL(scan_kernel, dim3(BB), dim3(256), 0, stream, ltgt, ends);
    hipLaunchKernelGGL(lr_expand_kernel, dim3(BB * MM), dim3(256), 0, stream,
                       enc, ends, comb, outbf);

    // 4) duration predictor on enc (T = 256)
    hipLaunchKernelGGL(conv_ln_relu_kernel, dim3(BB * (LPH / 16)), dim3(256), 0, stream,
                       encbf, wt[0], dp[1], dp[2], dp[3], h1, LPH);
    hipLaunchKernelGGL(conv_ln_relu_kernel, dim3(BB * (LPH / 16)), dim3(256), 0, stream,
                       h1, wt[1], dp[5], dp[6], dp[7], h2, LPH);
    hipLaunchKernelGGL(linear_relu_kernel, dim3((BB * LPH) / 8), dim3(256), 0, stream,
                       h2, dp[8], dp[9], durp);

    // 5) energy predictor on regulated output (T = 2048)
    hipLaunchKernelGGL(conv_ln_relu_kernel, dim3(BB * (MM / 16)), dim3(256), 0, stream,
                       outbf, wt[2], ep[1], ep[2], ep[3], h1, MM);
    hipLaunchKernelGGL(conv_ln_relu_kernel, dim3(BB * (MM / 16)), dim3(256), 0, stream,
                       h1, wt[3], ep[5], ep[6], ep[7], h2, MM);
    hipLaunchKernelGGL(linear_relu_kernel, dim3((BB * MM) / 8), dim3(256), 0, stream,
                       h2, ep[8], ep[9], enp);

    // 6) pitch predictor on regulated output (T = 2048)
    hipLaunchKernelGGL(conv_ln_relu_kernel, dim3(BB * (MM / 16)), dim3(256), 0, stream,
                       outbf, wt[4], pp[1], pp[2], pp[3], h1, MM);
    hipLaunchKernelGGL(conv_ln_relu_kernel, dim3(BB * (MM / 16)), dim3(256), 0, stream,
                       h1, wt[5], pp[5], pp[6], pp[7], h2, MM);
    hipLaunchKernelGGL(linear_relu_kernel, dim3((BB * MM) / 8), dim3(256), 0, stream,
                       h2, pp[8], pp[9], pip);

    // 7) bucketize embeddings + in-place add into combined
    hipLaunchKernelGGL(combine_kernel, dim3(BB * MM), dim3(256), 0, stream,
                       etgt, ptgt, eemb, pemb, comb);
}
